// GIN_Encoder_33397665693785
// MI455X (gfx1250) — matrix operand order repaired and checked by reading, not compiled, not run on hardware
//
#include <hip/hip_runtime.h>
#include <stdint.h>

// ---------------------------------------------------------------- types
typedef __attribute__((ext_vector_type(16))) __bf16 v16bf;
typedef __attribute__((ext_vector_type(8)))  float  v8f;

#define NNODES 100000
#define NEDGES 1600000
#define HID    128
#define LAT    64
#define LN_EPS 1e-5f
#define PADW   136   // bf16 elements per LDS row (stride padding, 16B-aligned rows)

union V16 { v16bf v; uint4 u[2]; };

__device__ __forceinline__ unsigned short f2bf(float f) {
  unsigned int u = __float_as_uint(f);
  u += 0x7fffu + ((u >> 16) & 1u);          // round-to-nearest-even
  return (unsigned short)(u >> 16);
}

// ------------------------------------------------- weight pre-swizzle (bf16, WMMA B-layout)
// dest flat index within a matrix: (((ntile*4 + kk)*32 + lane)*16 + v)
// element (K = kk*32 + (lane>>4)*8 + (v&7) + (v>>3)*16 , N = ntile*16 + (lane&15))
__global__ void prep_weights(const float* __restrict__ w0, const float* __restrict__ w1,
                             const float* __restrict__ w2, const float* __restrict__ w3,
                             const float* __restrict__ w4, const float* __restrict__ w5,
                             const float* __restrict__ l1, const float* __restrict__ l2,
                             unsigned short* __restrict__ packed) {
  int idx = blockIdx.x * 256 + threadIdx.x;
  if (idx >= 7 * 16384) return;
  int mat = idx >> 14, loc = idx & 16383;
  int v = loc & 15, lane = (loc >> 4) & 31, kk = (loc >> 9) & 3, nt = loc >> 11;
  int K = kk * 32 + (lane >> 4) * 8 + (v & 7) + (v >> 3) * 16;
  int N = nt * 16 + (lane & 15);
  float val;
  switch (mat) {
    case 0: val = w0[K * HID + N]; break;
    case 1: val = w1[K * HID + N]; break;
    case 2: val = w2[K * HID + N]; break;
    case 3: val = w3[K * HID + N]; break;
    case 4: val = w4[K * HID + N]; break;
    case 5: val = w5[K * HID + N]; break;
    default: val = (N < LAT) ? l1[K * LAT + N] : l2[K * LAT + (N - LAT)]; break;
  }
  packed[idx] = f2bf(val);
}

// ------------------------------------------------- agg init: agg = x  (GIN eps=0 self term)
__global__ void copy_f4(const float4* __restrict__ src, float4* __restrict__ dst, int n4) {
  int i = blockIdx.x * 256 + threadIdx.x;
  if (i < n4) dst[i] = src[i];
}

// ------------------------------------------------- edge scatter: agg[dst] += x[src]
// one wave32 per edge, 4 consecutive floats per lane (float4 gather, 4 f32 atomics)
__global__ void scatter_add(const float* __restrict__ x,
                            const long long* __restrict__ srcI,
                            const long long* __restrict__ dstI,
                            float* __restrict__ agg) {
  long long gtid = (long long)blockIdx.x * 256 + threadIdx.x;
  long long e = gtid >> 5;
  int lane = (int)(gtid & 31);
  if (e >= NEDGES) return;
  long long s = srcI[e];
  long long d = dstI[e];
  float4 xv = *(const float4*)(x + s * HID + lane * 4);
  float* ap = agg + d * HID + lane * 4;
  atomicAdd(ap + 0, xv.x);
  atomicAdd(ap + 1, xv.y);
  atomicAdd(ap + 2, xv.z);
  atomicAdd(ap + 3, xv.w);
}

// ------------------------------------------------- fused GIN MLP + LayerNorm
// block = 256 threads = 8 waves; wave w owns node rows [blk*128 + w*16, +16)
// h = LN(ReLU(ReLU(agg@W1+b1)@W2+b2)) * g + beta
__global__ __launch_bounds__(256) void gin_mlp_ln(
    const float* __restrict__ agg,
    const unsigned short* __restrict__ W1p,
    const unsigned short* __restrict__ W2p,
    const float* __restrict__ b1, const float* __restrict__ b2,
    const float* __restrict__ g,  const float* __restrict__ beta,
    float* __restrict__ hout) {
  __shared__ unsigned short htile[8 * 16 * PADW];   // per-wave 16 x 128 bf16 tiles (padded)

  int lane = threadIdx.x & 31;
  int wid  = threadIdx.x >> 5;
  int ln = lane & 15, hi = lane >> 4;
  int kb8 = hi * 8;
  int row0 = blockIdx.x * 128 + wid * 16;

  // ---- A tiles (16x32 bf16 per K-block) from agg, rows clamped for tail block ----
  int rowa = row0 + ln; if (rowa > NNODES - 1) rowa = NNODES - 1;
  const float* arow = agg + (long long)rowa * HID;
  V16 a[4];
#pragma unroll
  for (int kk = 0; kk < 4; kk++) {
    const float4* p0 = (const float4*)(arow + kk * 32 + kb8);       // K = kk*32+kb8 .. +7
    const float4* p1 = (const float4*)(arow + kk * 32 + 16 + kb8);  // K = kk*32+16+kb8 .. +7
    float4 f0 = p0[0], f1 = p0[1], f2 = p1[0], f3 = p1[1];
    a[kk].v[0]  = (__bf16)f0.x; a[kk].v[1]  = (__bf16)f0.y;
    a[kk].v[2]  = (__bf16)f0.z; a[kk].v[3]  = (__bf16)f0.w;
    a[kk].v[4]  = (__bf16)f1.x; a[kk].v[5]  = (__bf16)f1.y;
    a[kk].v[6]  = (__bf16)f1.z; a[kk].v[7]  = (__bf16)f1.w;
    a[kk].v[8]  = (__bf16)f2.x; a[kk].v[9]  = (__bf16)f2.y;
    a[kk].v[10] = (__bf16)f2.z; a[kk].v[11] = (__bf16)f2.w;
    a[kk].v[12] = (__bf16)f3.x; a[kk].v[13] = (__bf16)f3.y;
    a[kk].v[14] = (__bf16)f3.z; a[kk].v[15] = (__bf16)f3.w;
  }

  // ---- GEMM1: 8 N-tiles x 4 K-steps of v_wmma_f32_16x16x32_bf16 ----
  unsigned short* hb = htile + wid * 16 * PADW;
#pragma unroll
  for (int t = 0; t < 8; t++) {
    v8f c = {0.f, 0.f, 0.f, 0.f, 0.f, 0.f, 0.f, 0.f};
#pragma unroll
    for (int kk = 0; kk < 4; kk++) {
      V16 b;
      const uint4* bp = (const uint4*)(W1p + (((t * 4 + kk) * 32 + lane) << 4));
      b.u[0] = bp[0]; b.u[1] = bp[1];
      c = __builtin_amdgcn_wmma_f32_16x16x32_bf16(false, a[kk].v, false, b.v,
                                                  (short)0, c, false, false);
    }
    float bias = b1[t * 16 + ln];
#pragma unroll
    for (int r = 0; r < 8; r++) {   // D layout: lane holds (m = r + 8*hi, n = 16t + ln)
      float v = fmaxf(c[r] + bias, 0.f);
      hb[(r + 8 * hi) * PADW + (t * 16 + ln)] = f2bf(v);
    }
  }
  __syncthreads();

  // ---- reload intermediate in A-layout from LDS (ds_load_b128 runs) ----
  const unsigned short* hr = htile + wid * 16 * PADW + ln * PADW;
  V16 a2[4];
#pragma unroll
  for (int kk = 0; kk < 4; kk++) {
    a2[kk].u[0] = *(const uint4*)(hr + kk * 32 + kb8);
    a2[kk].u[1] = *(const uint4*)(hr + kk * 32 + 16 + kb8);
  }

  // ---- GEMM2 + bias + ReLU, keep full 16x128 wave tile in registers ----
  float vals[8][8];   // [t][r]
#pragma unroll
  for (int t = 0; t < 8; t++) {
    v8f c = {0.f, 0.f, 0.f, 0.f, 0.f, 0.f, 0.f, 0.f};
#pragma unroll
    for (int kk = 0; kk < 4; kk++) {
      V16 b;
      const uint4* bp = (const uint4*)(W2p + (((t * 4 + kk) * 32 + lane) << 4));
      b.u[0] = bp[0]; b.u[1] = bp[1];
      c = __builtin_amdgcn_wmma_f32_16x16x32_bf16(false, a2[kk].v, false, b.v,
                                                  (short)0, c, false, false);
    }
    float bias = b2[t * 16 + ln];
#pragma unroll
    for (int r = 0; r < 8; r++) vals[t][r] = fmaxf(c[r] + bias, 0.f);
  }

  // ---- LayerNorm: row m = r + 8*hi is striped over the 16-lane half + 8 t-tiles ----
  float s[8], q[8];
#pragma unroll
  for (int r = 0; r < 8; r++) {
    s[r] = 0.f; q[r] = 0.f;
#pragma unroll
    for (int t = 0; t < 8; t++) { s[r] += vals[t][r]; q[r] += vals[t][r] * vals[t][r]; }
  }
#pragma unroll
  for (int mask = 1; mask < 16; mask <<= 1) {
#pragma unroll
    for (int r = 0; r < 8; r++) {
      s[r] += __shfl_xor(s[r], mask, 32);
      q[r] += __shfl_xor(q[r], mask, 32);
    }
  }
  float mean[8], inv[8];
#pragma unroll
  for (int r = 0; r < 8; r++) {
    mean[r] = s[r] * (1.f / 128.f);
    float var = q[r] * (1.f / 128.f) - mean[r] * mean[r];
    inv[r] = rsqrtf(var + LN_EPS);
  }
#pragma unroll
  for (int t = 0; t < 8; t++) {
    int n = t * 16 + ln;
    float gn = g[n], bn = beta[n];
#pragma unroll
    for (int r = 0; r < 8; r++) {
      int row = row0 + r + 8 * hi;
      if (row < NNODES)
        hout[(long long)row * HID + n] = (vals[t][r] - mean[r]) * inv[r] * gn + bn;
    }
  }
}

// ------------------------------------------------- final: [mu|logvar] = ReLU(h3) @ [lin1|lin2] + b
__global__ __launch_bounds__(256) void final_proj(
    const float* __restrict__ h,
    const unsigned short* __restrict__ Wp,
    const float* __restrict__ l1b, const float* __restrict__ l2b,
    float* __restrict__ out) {
  int lane = threadIdx.x & 31;
  int wid  = threadIdx.x >> 5;
  int ln = lane & 15, hi = lane >> 4;
  int kb8 = hi * 8;
  int row0 = blockIdx.x * 128 + wid * 16;

  int rowa = row0 + ln; if (rowa > NNODES - 1) rowa = NNODES - 1;
  const float* arow = h + (long long)rowa * HID;
  V16 a[4];
#pragma unroll
  for (int kk = 0; kk < 4; kk++) {
    const float4* p0 = (const float4*)(arow + kk * 32 + kb8);
    const float4* p1 = (const float4*)(arow + kk * 32 + 16 + kb8);
    float4 f0 = p0[0], f1 = p0[1], f2 = p1[0], f3 = p1[1];
    a[kk].v[0]  = (__bf16)fmaxf(f0.x, 0.f); a[kk].v[1]  = (__bf16)fmaxf(f0.y, 0.f);
    a[kk].v[2]  = (__bf16)fmaxf(f0.z, 0.f); a[kk].v[3]  = (__bf16)fmaxf(f0.w, 0.f);
    a[kk].v[4]  = (__bf16)fmaxf(f1.x, 0.f); a[kk].v[5]  = (__bf16)fmaxf(f1.y, 0.f);
    a[kk].v[6]  = (__bf16)fmaxf(f1.z, 0.f); a[kk].v[7]  = (__bf16)fmaxf(f1.w, 0.f);
    a[kk].v[8]  = (__bf16)fmaxf(f2.x, 0.f); a[kk].v[9]  = (__bf16)fmaxf(f2.y, 0.f);
    a[kk].v[10] = (__bf16)fmaxf(f2.z, 0.f); a[kk].v[11] = (__bf16)fmaxf(f2.w, 0.f);
    a[kk].v[12] = (__bf16)fmaxf(f3.x, 0.f); a[kk].v[13] = (__bf16)fmaxf(f3.y, 0.f);
    a[kk].v[14] = (__bf16)fmaxf(f3.z, 0.f); a[kk].v[15] = (__bf16)fmaxf(f3.w, 0.f);
  }

#pragma unroll
  for (int t = 0; t < 8; t++) {
    v8f c = {0.f, 0.f, 0.f, 0.f, 0.f, 0.f, 0.f, 0.f};
#pragma unroll
    for (int kk = 0; kk < 4; kk++) {
      V16 b;
      const uint4* bp = (const uint4*)(Wp + (((t * 4 + kk) * 32 + lane) << 4));
      b.u[0] = bp[0]; b.u[1] = bp[1];
      c = __builtin_amdgcn_wmma_f32_16x16x32_bf16(false, a[kk].v, false, b.v,
                                                  (short)0, c, false, false);
    }
    int n = t * 16 + ln;
    float bias = (n < LAT) ? l1b[n] : l2b[n - LAT];
#pragma unroll
    for (int r = 0; r < 8; r++) {
      int row = row0 + r + 8 * hi;
      if (row < NNODES) {
        float v = c[r] + bias;
        if (n < LAT) out[(long long)row * LAT + n] = v;                            // mu
        else out[(long long)NNODES * LAT + (long long)row * LAT + (n - LAT)] = v;  // logvar
      }
    }
  }
}

// ================================================= host orchestration
extern "C" void kernel_launch(void* const* d_in, const int* in_sizes, int n_in,
                              void* d_out, int out_size, void* d_ws, size_t ws_size,
                              hipStream_t stream) {
  const float*     x   = (const float*)d_in[0];
  const long long* ei  = (const long long*)d_in[1];   // int64 [2, NEDGES]
  const float* cW1[3] = {(const float*)d_in[2],  (const float*)d_in[8],  (const float*)d_in[14]};
  const float* cb1[3] = {(const float*)d_in[3],  (const float*)d_in[9],  (const float*)d_in[15]};
  const float* cW2[3] = {(const float*)d_in[4],  (const float*)d_in[10], (const float*)d_in[16]};
  const float* cb2[3] = {(const float*)d_in[5],  (const float*)d_in[11], (const float*)d_in[17]};
  const float* cg [3] = {(const float*)d_in[6],  (const float*)d_in[12], (const float*)d_in[18]};
  const float* cbe[3] = {(const float*)d_in[7],  (const float*)d_in[13], (const float*)d_in[19]};
  const float* l1W = (const float*)d_in[20];
  const float* l1b = (const float*)d_in[21];
  const float* l2W = (const float*)d_in[22];
  const float* l2b = (const float*)d_in[23];

  // workspace carve (both node buffers fit in the 192MB L2)
  char* ws = (char*)d_ws;
  float* aggb = (float*)ws;                                   // 100000*128*4 = 51.2 MB
  float* hbuf = (float*)(ws + 51200000ULL);                   // 51.2 MB
  unsigned short* packed = (unsigned short*)(ws + 102400000ULL); // 7 * 32 KB bf16 swizzled

  const int GB = (NNODES + 127) / 128;          // 782 blocks of 8 waves x 16 rows
  const int C4 = (NNODES * HID / 4 + 255) / 256;

  prep_weights<<<(7 * 16384 + 255) / 256, 256, 0, stream>>>(
      cW1[0], cW2[0], cW1[1], cW2[1], cW1[2], cW2[2], l1W, l2W, packed);

  const float* cur = x;
  for (int L = 0; L < 3; L++) {
    copy_f4<<<C4, 256, 0, stream>>>((const float4*)cur, (float4*)aggb, NNODES * HID / 4);
    scatter_add<<<NEDGES / 8, 256, 0, stream>>>(cur, ei, ei + NEDGES, aggb);
    gin_mlp_ln<<<GB, 256, 0, stream>>>(aggb,
                                       packed + (2 * L + 0) * 16384,
                                       packed + (2 * L + 1) * 16384,
                                       cb1[L], cb2[L], cg[L], cbe[L], hbuf);
    cur = hbuf;
  }
  final_proj<<<GB, 256, 0, stream>>>(hbuf, packed + 6 * 16384, l1b, l2b, (float*)d_out);
}
